// GrpahScalarRegressor_35716948033579
// MI455X (gfx1250) — matrix operand order repaired and checked by reading, not compiled, not run on hardware
//
#include <hip/hip_runtime.h>
#include <hip/hip_bf16.h>

typedef __attribute__((ext_vector_type(16))) __bf16 v16bf;
typedef __attribute__((ext_vector_type(8)))  float  v8f;

#define EPSF 1e-5f

__device__ __forceinline__ unsigned short f2bf(float f){
  unsigned u = __float_as_uint(f);
  return (unsigned short)((u + 0x7FFFu + ((u >> 16) & 1u)) >> 16);
}

// A-matrix 16x32 bf16 fragment: lane half h: elems 0..15 = K 16h+0..15 (contiguous)
__device__ __forceinline__ v16bf fragA(const unsigned short* p){
  v16bf r;
  ((float4*)&r)[0] = *(const float4*)(p);
  ((float4*)&r)[1] = *(const float4*)(p + 8);
  return r;
}
// B-matrix 32x16 bf16 fragment: lane half h: elems 0..15 = K 16h+0..15 (contiguous)
__device__ __forceinline__ v16bf fragB(const unsigned short* p){
  v16bf r;
  ((float4*)&r)[0] = *(const float4*)(p);
  ((float4*)&r)[1] = *(const float4*)(p + 8);
  return r;
}

__global__ void k_fill(float* p, float v, int n){
  int i = blockIdx.x * blockDim.x + threadIdx.x;
  if (i < n) p[i] = v;
}

__global__ void k_deg(const int* __restrict__ dst, float* deg, int E){
  int e = blockIdx.x * blockDim.x + threadIdx.x;
  if (e < E) atomicAdd(deg + dst[e], 1.0f);
}

__global__ void k_dinv(float* dinv, float* self, int N){
  int n = blockIdx.x * blockDim.x + threadIdx.x;
  if (n >= N) return;
  float dv = rsqrtf(dinv[n]);   // dinv currently holds deg
  dinv[n] = dv;
  self[n] = dv * dv;
}

__global__ void k_ecoef(const int* __restrict__ src, const int* __restrict__ dst,
                        const float* __restrict__ dinv, float* ecoef, int E){
  int e = blockIdx.x * blockDim.x + threadIdx.x;
  if (e < E) ecoef[e] = dinv[src[e]] * dinv[dst[e]];
}

__global__ void k_wstats(const float* __restrict__ w, float* stat, int N){
  __shared__ float s0[256], s1[256];
  int i = blockIdx.x * 256 + threadIdx.x;
  float a = 0.f, b = 0.f;
  if (i < N){ a = w[i]; b = a * a; }
  s0[threadIdx.x] = a; s1[threadIdx.x] = b; __syncthreads();
  for (int o = 128; o > 0; o >>= 1){
    if (threadIdx.x < o){ s0[threadIdx.x] += s0[threadIdx.x + o]; s1[threadIdx.x] += s1[threadIdx.x + o]; }
    __syncthreads();
  }
  if (threadIdx.x == 0){ atomicAdd(stat, s0[0]); atomicAdd(stat + 1, s1[0]); }
}

// column-wise sum & sum of squares over N rows
__global__ void k_colstats(const float* __restrict__ m, float* colsum, float* colsumsq, int N, int D){
  int d = blockIdx.x * blockDim.x + threadIdx.x;
  if (d >= D) return;
  float s = 0.f, s2 = 0.f;
  for (int n = blockIdx.y; n < N; n += gridDim.y){
    float v = m[(size_t)n * D + d];
    s += v; s2 += v * v;
  }
  atomicAdd(colsum + d, s);
  atomicAdd(colsumsq + d, s2);
}

__global__ void k_bnfinal(const float* __restrict__ cs, const float* __restrict__ css,
                          const float* __restrict__ g, const float* __restrict__ b,
                          float* scale, float* shift, int D, float invN){
  int d = blockIdx.x * blockDim.x + threadIdx.x;
  if (d >= D) return;
  float mean = cs[d] * invN;
  float var  = css[d] * invN - mean * mean;
  float s = g[d] * rsqrtf(var + EPSF);
  scale[d] = s;
  shift[d] = b[d] - mean * s;
}

__global__ void k_h0(const float* __restrict__ x, const float* __restrict__ wts,
                     const float* __restrict__ scaleX, const float* __restrict__ shiftX,
                     const float* __restrict__ wstat, const float* __restrict__ embedW,
                     const float* __restrict__ bnw_g, const float* __restrict__ bnw_b,
                     unsigned short* hbf, int N, int D, float invN){
  int i = blockIdx.x * blockDim.x + threadIdx.x;
  if (i >= N * D) return;
  int n = i / D, d = i - n * D;
  float mw = wstat[0] * invN;
  float vw = wstat[1] * invN - mw * mw;
  float ew = embedW[d];
  float cw = ew * rsqrtf(vw * ew * ew + EPSF) * bnw_g[d];
  float h = x[i] * scaleX[d] + shiftX[d] + (wts[n] - mw) * cw + bnw_b[d];
  hbf[i] = f2bf(fmaxf(h, 0.f));
}

__global__ void k_prepW(const float* __restrict__ W, unsigned short* Wt, int total, int D){
  int i = blockIdx.x * blockDim.x + threadIdx.x;
  if (i >= total) return;
  int l = i / (D * D); int rem = i - l * D * D;
  int n = rem / D;     int k = rem - n * D;
  Wt[i] = f2bf(W[(size_t)l * D * D + (size_t)k * D + n]);  // transpose: Wt[l][n][k] = W[l][k][n]
}

// y[N,D] = hbf[N,D] @ W  (W pre-transposed bf16), one wave -> 16x64 output strip
__global__ void k_gemm(const unsigned short* __restrict__ hb, const unsigned short* __restrict__ Wt,
                       float* __restrict__ y, int N, int D){
  int wid  = (blockIdx.x * blockDim.x + threadIdx.x) >> 5;
  int lane = threadIdx.x & 31;
  int ncg  = D >> 6;                    // 64-wide column groups
  int tileRow = wid / ncg;
  int cg = wid - tileRow * ncg;
  if (tileRow * 16 >= N) return;
  int half = lane >> 4;
  int mr   = lane & 15;
  size_t rowBase = (size_t)tileRow * 16;
  int colBase = cg * 64;
  const unsigned short* ap = hb + (rowBase + mr) * (size_t)D + 16 * half;
  const unsigned short* bp = Wt + (size_t)(colBase + mr) * D + 16 * half;
  size_t bstride = (size_t)16 * D;
  v8f c0 = {}, c1 = {}, c2 = {}, c3 = {};
  #pragma unroll 4
  for (int kk = 0; kk < D; kk += 32){
    v16bf a  = fragA(ap + kk);
    v16bf b0 = fragB(bp + kk);
    v16bf b1 = fragB(bp + bstride + kk);
    v16bf b2 = fragB(bp + 2 * bstride + kk);
    v16bf b3 = fragB(bp + 3 * bstride + kk);
    c0 = __builtin_amdgcn_wmma_f32_16x16x32_bf16(false, a, false, b0, (short)0, c0, false, false);
    c1 = __builtin_amdgcn_wmma_f32_16x16x32_bf16(false, a, false, b1, (short)0, c1, false, false);
    c2 = __builtin_amdgcn_wmma_f32_16x16x32_bf16(false, a, false, b2, (short)0, c2, false, false);
    c3 = __builtin_amdgcn_wmma_f32_16x16x32_bf16(false, a, false, b3, (short)0, c3, false, false);
  }
  float* yr = y + (rowBase + 8 * half) * D + colBase + mr;
  #pragma unroll
  for (int r = 0; r < 8; ++r){
    yr[(size_t)r * D +  0] = c0[r];
    yr[(size_t)r * D + 16] = c1[r];
    yr[(size_t)r * D + 32] = c2[r];
    yr[(size_t)r * D + 48] = c3[r];
  }
}

__global__ void k_selfinit(const float4* __restrict__ y, const float* __restrict__ self,
                           const float* __restrict__ bias, float4* agg, int N, int D4){
  int i = blockIdx.x * blockDim.x + threadIdx.x;
  if (i >= N * D4) return;
  int n = i / D4; int d4 = i - n * D4;
  float c = self[n];
  float4 v = y[i];
  const float* b = bias + d4 * 4;
  float4 r; r.x = v.x * c + b[0]; r.y = v.y * c + b[1]; r.z = v.z * c + b[2]; r.w = v.w * c + b[3];
  agg[i] = r;
}

__global__ void k_scatter(const float4* __restrict__ y, const int* __restrict__ src,
                          const int* __restrict__ dst, const float* __restrict__ ecoef,
                          float* agg, int E, int D4){
  int t = blockIdx.x * blockDim.x + threadIdx.x;
  if (t >= E * D4) return;
  int e = t / D4; int d4 = t - e * D4;
  float c = ecoef[e];
  float4 v = y[(size_t)src[e] * D4 + d4];
  float* a = agg + ((size_t)dst[e] * D4 + d4) * 4;
  atomicAdd(a + 0, v.x * c);
  atomicAdd(a + 1, v.y * c);
  atomicAdd(a + 2, v.z * c);
  atomicAdd(a + 3, v.w * c);
}

__global__ void k_relu_bf(const float* __restrict__ a, const float* __restrict__ scale,
                          const float* __restrict__ shift, unsigned short* hbf, int N, int D){
  int i = blockIdx.x * blockDim.x + threadIdx.x;
  if (i >= N * D) return;
  int d = i % D;
  float v = a[i] * scale[d] + shift[d];
  hbf[i] = f2bf(fmaxf(v, 0.f));
}

__global__ void k_pool(const float4* __restrict__ agg, const int* __restrict__ batch,
                       float* pooled, int N, int D4){
  int i = blockIdx.x * blockDim.x + threadIdx.x;
  if (i >= N * D4) return;
  int n = i / D4; int d4 = i - n * D4;
  float4 v = agg[i];
  float* p = pooled + (size_t)batch[n] * (D4 * 4) + d4 * 4;
  atomicAdd(p + 0, v.x); atomicAdd(p + 1, v.y); atomicAdd(p + 2, v.z); atomicAdd(p + 3, v.w);
}

__global__ void k_cnt(const int* __restrict__ batch, float* cnt, int N){
  int n = blockIdx.x * blockDim.x + threadIdx.x;
  if (n < N) atomicAdd(cnt + batch[n], 1.0f);
}

__global__ void k_final(const float* __restrict__ pooled, const float* __restrict__ cnt,
                        const float* __restrict__ linW, const float* __restrict__ linB,
                        float* out, int D){
  __shared__ float sm[128];
  int g = blockIdx.x, tid = threadIdx.x;
  float s = 0.f;
  for (int d = tid; d < D; d += 128) s += pooled[(size_t)g * D + d] * linW[d];
  sm[tid] = s; __syncthreads();
  for (int o = 64; o > 0; o >>= 1){
    if (tid < o) sm[tid] += sm[tid + o];
    __syncthreads();
  }
  if (tid == 0) out[g] = sm[0] / fmaxf(cnt[g], 1.0f) + linB[0];
}

static inline int cdiv(long long a, int b){ return (int)((a + b - 1) / b); }

extern "C" void kernel_launch(void* const* d_in, const int* in_sizes, int n_in,
                              void* d_out, int out_size, void* d_ws, size_t ws_size,
                              hipStream_t stream) {
  const float* x       = (const float*)d_in[0];
  const float* wts     = (const float*)d_in[1];
  const int*   ei      = (const int*)  d_in[2];
  const int*   batch   = (const int*)  d_in[3];
  const float* embedW  = (const float*)d_in[4];
  // d_in[5] = embedB (cancels inside BN of ew)
  const float* bn0_g   = (const float*)d_in[6];
  const float* bn0_b   = (const float*)d_in[7];
  const float* bnw_g   = (const float*)d_in[8];
  const float* bnw_b   = (const float*)d_in[9];
  const float* convW   = (const float*)d_in[10];
  const float* convB   = (const float*)d_in[11];
  const float* bns_g   = (const float*)d_in[12];
  const float* bns_b   = (const float*)d_in[13];
  const float* linW    = (const float*)d_in[14];
  const float* linB    = (const float*)d_in[15];
  float* out = (float*)d_out;

  const int N = in_sizes[1];
  const int E = in_sizes[2] / 2;
  const int D = in_sizes[5];
  const int G = out_size;
  const int L = in_sizes[10] / (D * D);
  const int D4 = D / 4;
  const float invN = 1.0f / (float)N;
  const int* src = ei;
  const int* dst = ei + E;

  // workspace carve-up
  char* w = (char*)d_ws;
  auto alloc = [&](size_t bytes) -> void* {
    void* p = (void*)w; w += (bytes + 255) & ~(size_t)255; return p;
  };
  float* dinv    = (float*)alloc((size_t)N * 4);       // deg, then 1/sqrt(deg)
  float* self    = (float*)alloc((size_t)N * 4);
  float* ecoef   = (float*)alloc((size_t)E * 4);
  float* wstat   = (float*)alloc(256);
  float* colsum  = (float*)alloc((size_t)2 * D * 4);   // colsum | colsumsq contiguous
  float* colsumsq = colsum + D;
  float* scale   = (float*)alloc((size_t)D * 4);
  float* shift   = (float*)alloc((size_t)D * 4);
  float* pooled  = (float*)alloc((size_t)G * D * 4);
  float* cnt     = (float*)alloc((size_t)G * 4);
  unsigned short* hbf = (unsigned short*)alloc((size_t)N * D * 2);
  unsigned short* Wt  = (unsigned short*)alloc((size_t)L * D * D * 2);
  float* y       = (float*)alloc((size_t)N * D * 4);
  float* agg     = (float*)alloc((size_t)N * D * 4);
  (void)ws_size; (void)n_in;

  const int T = 256;

  // --- GCN normalization coefficients ---
  k_fill<<<cdiv(N, T), T, 0, stream>>>(dinv, 1.0f, N);            // self-loop
  k_deg <<<cdiv(E, T), T, 0, stream>>>(dst, dinv, E);
  k_dinv<<<cdiv(N, T), T, 0, stream>>>(dinv, self, N);
  k_ecoef<<<cdiv(E, T), T, 0, stream>>>(src, dst, dinv, ecoef, E);

  // --- input batch norms + first activation (bf16) ---
  k_fill<<<1, T, 0, stream>>>(wstat, 0.0f, 2);
  k_wstats<<<cdiv(N, T), T, 0, stream>>>(wts, wstat, N);
  k_fill<<<cdiv(2 * D, T), T, 0, stream>>>(colsum, 0.0f, 2 * D);
  {
    dim3 g(cdiv(D, T), 64);
    k_colstats<<<g, T, 0, stream>>>(x, colsum, colsumsq, N, D);
  }
  k_bnfinal<<<cdiv(D, T), T, 0, stream>>>(colsum, colsumsq, bn0_g, bn0_b, scale, shift, D, invN);
  k_h0<<<cdiv((long long)N * D, T), T, 0, stream>>>(x, wts, scale, shift, wstat, embedW,
                                                    bnw_g, bnw_b, hbf, N, D, invN);

  // --- pre-transpose conv weights to bf16 ---
  k_prepW<<<cdiv((long long)L * D * D, T), T, 0, stream>>>(convW, Wt, L * D * D, D);

  // --- GCN layers ---
  const int nWaves = cdiv(N, 16) * (D / 64);
  const int gemmBlocks = cdiv((long long)nWaves * 32, T);
  for (int l = 0; l < L; ++l){
    k_gemm<<<gemmBlocks, T, 0, stream>>>(hbf, Wt + (size_t)l * D * D, y, N, D);
    k_selfinit<<<cdiv((long long)N * D4, T), T, 0, stream>>>(
        (const float4*)y, self, convB + (size_t)l * D, (float4*)agg, N, D4);
    k_scatter<<<cdiv((long long)E * D4, T), T, 0, stream>>>(
        (const float4*)y, src, dst, ecoef, agg, E, D4);
    if (l < L - 1){
      k_fill<<<cdiv(2 * D, T), T, 0, stream>>>(colsum, 0.0f, 2 * D);
      dim3 g(cdiv(D, T), 64);
      k_colstats<<<g, T, 0, stream>>>(agg, colsum, colsumsq, N, D);
      k_bnfinal<<<cdiv(D, T), T, 0, stream>>>(colsum, colsumsq,
                                              bns_g + (size_t)l * D, bns_b + (size_t)l * D,
                                              scale, shift, D, invN);
      k_relu_bf<<<cdiv((long long)N * D, T), T, 0, stream>>>(agg, scale, shift, hbf, N, D);
    }
  }

  // --- global mean pool + linear head ---
  k_fill<<<cdiv(G * D, T), T, 0, stream>>>(pooled, 0.0f, G * D);
  k_fill<<<1, T, 0, stream>>>(cnt, 0.0f, G);
  k_pool<<<cdiv((long long)N * D4, T), T, 0, stream>>>((const float4*)agg, batch, pooled, N, D4);
  k_cnt<<<cdiv(N, T), T, 0, stream>>>(batch, cnt, N);
  k_final<<<G, 128, 0, stream>>>(pooled, cnt, linW, linB, out, D);
}
